// Mamba_Neck_14972255994045
// MI455X (gfx1250) — matrix operand order repaired and checked by reading, not compiled, not run on hardware
//
#include <hip/hip_runtime.h>
#include <hip/hip_bf16.h>

// ---------------- model dims ----------------
#define B_  1
#define NX_ 256
#define L_  1024
#define C_  512
#define DI_ 1024
#define DS_ 16
#define DR_ 32
#define NL_ 4
#define H_  8
#define DH_ 64
#define FH_ 128

typedef __attribute__((ext_vector_type(16))) _Float16 v16h;
typedef __attribute__((ext_vector_type(8)))  float    v8f;

// ---------------- wave32 reductions ----------------
__device__ __forceinline__ float wave_sum(float v) {
  for (int m = 16; m; m >>= 1) v += __shfl_xor(v, m, 32);
  return v;
}
__device__ __forceinline__ float wave_max(float v) {
  for (int m = 16; m; m >>= 1) v = fmaxf(v, __shfl_xor(v, m, 32));
  return v;
}

__device__ __forceinline__ float act_apply(float v, int act) {
  if (act == 1) return v / (1.f + __expf(-v));                       // silu
  if (act == 2) return 0.5f * v * (1.f + erff(v * 0.70710678118f));  // gelu exact
  if (act == 3) return (v > 20.f) ? v : log1pf(__expf(v));           // softplus
  return v;
}

// ---------------- WMMA GEMM: C[M,N] = A[M,K] * op(B) (+bias)(+act) ----------------
// One wave computes a 16x64 tile: 4 chained v_wmma_f32_16x16x32_f16 accumulators
// reuse a single A fragment per k-step.
// A: f32 row-major, lda.  B: f32.
//   bT==0 -> B is [N,K] row-major (x @ W^T): direct contiguous fragment loads
//            + global_prefetch of the next k-step.
//   bT==1 -> B is [K,N] row-major (P @ V): 32x64 f32 tile copied into LDS with
//            GLOBAL_LOAD_ASYNC_TO_LDS_B128 (one 256B row per lane, 16B/lane/op),
//            completion via s_wait_asynccnt, fragments read strided from LDS
//            with f32->f16 convert at read time.
// Grid: (N/64, M/16), block 32.  K multiple of 32, N multiple of 64 (all calls).
__global__ void gemm_wmma_k(const float* __restrict__ A, int lda,
                            const float* __restrict__ Bm, int ldb, int bT,
                            float* __restrict__ Cm, int ldc, int K,
                            const float* __restrict__ bias, int act) {
  __shared__ float lbf[32 * 64];     // [k][n] f32 B tile (8 KB)
  const int lane = threadIdx.x & 31;
  const int l16  = lane & 15;
  const int hh   = lane >> 4;
  const int m0  = blockIdx.y * 16;
  const int n0w = blockIdx.x * 64;

  v8f acc[4] = {};
  for (int k0 = 0; k0 < K; k0 += 32) {
    // ---- A fragment (16x32 f16): lane = row m0+l16, K chunk [k0+hh*16,+16)
    v16h a;
    const float* arow = A + (size_t)(m0 + l16) * lda + k0 + hh * 16;
#pragma unroll
    for (int t = 0; t < 8; ++t) a[t] = (_Float16)arow[t];
#pragma unroll
    for (int t = 0; t < 8; ++t) a[8 + t] = (_Float16)arow[8 + t];
    if (k0 + 32 < K) __builtin_prefetch(arow + 32, 0, 1);  // global_prefetch next k-step

    if (bT) {
      // ---- async-copy B 32x64 f32 tile into LDS: lane owns row k = k0+lane
      __syncthreads();  // WAR guard vs previous iteration's fragment reads
      const float* grow = Bm + (size_t)(k0 + lane) * ldb + n0w;   // 256B contiguous
      unsigned ldsoff = (unsigned)(uintptr_t)(&lbf[lane * 64]);   // per-lane LDS row
#pragma unroll
      for (int c = 0; c < 16; ++c) {   // 16 x 16B per lane = 256B row
        asm volatile("global_load_async_to_lds_b128 %0, %1, off"
                     :: "v"(ldsoff), "v"(grow) : "memory");
        ldsoff += 16;
        grow   += 4;
      }
      asm volatile("s_wait_asynccnt 0" ::: "memory");
      __syncthreads();
    }

#pragma unroll
    for (int u = 0; u < 4; ++u) {
      // ---- B fragment (32x16 f16): lane = col (u*16+l16), K chunk [k0+hh*16,+16)
      v16h b;
      if (!bT) {
        const float* brow = Bm + (size_t)(n0w + u * 16 + l16) * ldb + k0 + hh * 16;
#pragma unroll
        for (int t = 0; t < 16; ++t) b[t] = (_Float16)brow[t];
        if (k0 + 32 < K) __builtin_prefetch(brow + 32, 0, 1);
      } else {
        const float* lcol = lbf + (hh * 16) * 64 + (u * 16 + l16);
#pragma unroll
        for (int t = 0; t < 16; ++t) b[t] = (_Float16)lcol[t * 64];
      }
      acc[u] = __builtin_amdgcn_wmma_f32_16x16x32_f16(false, a, false, b, (short)0,
                                                      acc[u], false, false);
    }
  }
#pragma unroll
  for (int u = 0; u < 4; ++u) {
    const int n = n0w + u * 16 + l16;
    const float bv = bias ? bias[n] : 0.f;
#pragma unroll
    for (int r = 0; r < 8; ++r) {
      const int m = m0 + r + 8 * hh;
      Cm[(size_t)m * ldc + n] = act_apply(acc[u][r] + bv, act);
    }
  }
}

// ---------------- LayerNorm / RMSNorm: one wave per row ----------------
__global__ void ln_k(const float* __restrict__ x, const float* __restrict__ w,
                     const float* __restrict__ b, float* __restrict__ y, int Cn) {
  const int row = blockIdx.x, lane = threadIdx.x;
  const float* xr = x + (size_t)row * Cn;
  float s = 0.f;
  for (int c = lane; c < Cn; c += 32) s += xr[c];
  const float mean = wave_sum(s) / Cn;
  float v = 0.f;
  for (int c = lane; c < Cn; c += 32) { float d = xr[c] - mean; v += d * d; }
  const float inv = rsqrtf(wave_sum(v) / Cn + 1e-6f);
  for (int c = lane; c < Cn; c += 32)
    y[(size_t)row * Cn + c] = (xr[c] - mean) * inv * w[c] + b[c];
}

__global__ void rms_k(const float* __restrict__ x, const float* __restrict__ w,
                      float* __restrict__ y, int Cn) {
  const int row = blockIdx.x, lane = threadIdx.x;
  const float* xr = x + (size_t)row * Cn;
  float s = 0.f;
  for (int c = lane; c < Cn; c += 32) s += xr[c] * xr[c];
  const float inv = rsqrtf(wave_sum(s) / Cn + 1e-5f);
  for (int c = lane; c < Cn; c += 32)
    y[(size_t)row * Cn + c] = xr[c] * inv * w[c];
}

// ---------------- row softmax(S*scale), in place; one wave per row ----------------
__global__ void softmax_row_k(float* __restrict__ S, int N, float scale) {
  const int row = blockIdx.x, lane = threadIdx.x;
  float* r = S + (size_t)row * N;
  float mx = -3.4e38f;
  for (int c = lane; c < N; c += 32) mx = fmaxf(mx, r[c] * scale);
  mx = wave_max(mx);
  float sum = 0.f;
  for (int c = lane; c < N; c += 32) { float e = __expf(r[c] * scale - mx); r[c] = e; sum += e; }
  const float inv = 1.f / wave_sum(sum);
  for (int c = lane; c < N; c += 32) r[c] *= inv;
}

// ---------------- depthwise centered conv1d (k=3, pad 1) + SiLU ----------------
__global__ void conv1d_silu_k(const float* __restrict__ xin, int ldx,
                              const float* __restrict__ w, const float* __restrict__ bc,
                              float* __restrict__ out) {
  int idx = blockIdx.x * blockDim.x + threadIdx.x;
  if (idx >= L_ * DI_) return;
  const int l = idx / DI_, d = idx % DI_;
  float acc = bc[d];
#pragma unroll
  for (int j = 0; j < 3; ++j) {
    const int t = l - 1 + j;
    if (t >= 0 && t < L_) acc += xin[(size_t)t * ldx + d] * w[d * 3 + j];
  }
  out[idx] = acc / (1.f + __expf(-acc));
}

// ---------------- selective state update (single step, elementwise over s) ----------------
__global__ void scan_k(const float* __restrict__ XC, const float* __restrict__ DL,
                       const float* __restrict__ DBC, const float* __restrict__ A_log,
                       const float* __restrict__ Dv, const float* __restrict__ h_in,
                       float* __restrict__ h_out, float* __restrict__ Y) {
  int idx = blockIdx.x * blockDim.x + threadIdx.x;
  if (idx >= L_ * DI_) return;
  const int l = idx / DI_, d = idx % DI_;
  const float xc = XC[idx], dl = DL[idx], dx = dl * xc;
  const float* Bm = DBC + (size_t)l * 64 + 32;
  const float* Cm = DBC + (size_t)l * 64 + 48;
  const float* hi = h_in + (size_t)idx * DS_;
  float* ho = h_out + (size_t)idx * DS_;
  float y = 0.f;
#pragma unroll
  for (int s = 0; s < DS_; ++s) {
    const float a = -__expf(A_log[d * DS_ + s]);
    const float hn = __expf(dl * a) * hi[s] + dx * Bm[s];
    ho[s] = hn;
    y += hn * Cm[s];
  }
  Y[idx] = y + Dv[d] * xc;
}

// ---------------- y_gated = silu(z) * y ----------------
__global__ void gate_silu_k(const float* __restrict__ Z, int ldz,
                            const float* __restrict__ Y, float* __restrict__ YG) {
  int idx = blockIdx.x * blockDim.x + threadIdx.x;
  if (idx >= L_ * DI_) return;
  const int l = idx / DI_, d = idx % DI_;
  const float z = Z[(size_t)l * ldz + d];
  YG[idx] = (z / (1.f + __expf(-z))) * Y[idx];
}

// ---------------- elementwise add / gamma-scaled add ----------------
__global__ void add_k(const float* __restrict__ a, const float* __restrict__ b,
                      float* __restrict__ y, int n) {
  int idx = blockIdx.x * blockDim.x + threadIdx.x;
  if (idx < n) y[idx] = a[idx] + b[idx];
}
__global__ void axpy_col_k(float* __restrict__ X, const float* __restrict__ G,
                           const float* __restrict__ A, int n, int Cn) {
  int idx = blockIdx.x * blockDim.x + threadIdx.x;
  if (idx < n) X[idx] += G[idx % Cn] * A[idx];
}

// ---------------- 3x3 depthwise conv on 32x32 token grid + bias + GELU ----------------
__global__ void dw3x3_gelu_k(const float* __restrict__ T, const float* __restrict__ W,
                             const float* __restrict__ Bb, float* __restrict__ O) {
  int idx = blockIdx.x * blockDim.x + threadIdx.x;
  if (idx >= L_ * FH_) return;
  const int ch = idx % FH_, n = idx / FH_;
  const int i = n >> 5, j = n & 31;
  float acc = Bb[ch];
#pragma unroll
  for (int di = -1; di <= 1; ++di)
#pragma unroll
    for (int dj = -1; dj <= 1; ++dj) {
      const int ii = i + di, jj = j + dj;
      if (ii >= 0 && ii < 32 && jj >= 0 && jj < 32)
        acc += T[(size_t)((ii << 5) + jj) * FH_ + ch] * W[ch * 9 + (di + 1) * 3 + (dj + 1)];
    }
  O[idx] = 0.5f * acc * (1.f + erff(acc * 0.70710678118f));
}

// =====================================================================
extern "C" void kernel_launch(void* const* d_in, const int* in_sizes, int n_in,
                              void* d_out, int out_size, void* d_ws, size_t ws_size,
                              hipStream_t stream) {
  (void)in_sizes; (void)n_in; (void)out_size; (void)ws_size;
  // ---- inputs (setup_inputs order) ----
  const float* x_in    = (const float*)d_in[0];
  const float* xs_in   = (const float*)d_in[1];
  const float* h_in    = (const float*)d_in[2];
  const float* m_in_w  = (const float*)d_in[3];
  const float* m_conv_w= (const float*)d_in[4];
  const float* m_conv_b= (const float*)d_in[5];
  const float* m_xp_w  = (const float*)d_in[6];
  const float* m_dt_w  = (const float*)d_in[7];
  const float* m_dt_b  = (const float*)d_in[8];
  const float* m_A_log = (const float*)d_in[9];
  const float* m_D     = (const float*)d_in[10];
  const float* m_out_w = (const float*)d_in[11];
  const float* m_norm_w= (const float*)d_in[12];
  const float* i_qn_w  = (const float*)d_in[13];
  const float* i_qn_b  = (const float*)d_in[14];
  const float* i_fn_w  = (const float*)d_in[15];
  const float* i_fn_b  = (const float*)d_in[16];
  const float* i_in_w  = (const float*)d_in[17];
  const float* i_in_b  = (const float*)d_in[18];
  const float* i_out_w = (const float*)d_in[19];
  const float* i_out_b = (const float*)d_in[20];
  const float* i_gamma = (const float*)d_in[21];
  const float* e_qn_w  = (const float*)d_in[22];
  const float* e_qn_b  = (const float*)d_in[23];
  const float* e_fn_w  = (const float*)d_in[24];
  const float* e_fn_b  = (const float*)d_in[25];
  const float* e_in_w  = (const float*)d_in[26];
  const float* e_in_b  = (const float*)d_in[27];
  const float* e_out_w = (const float*)d_in[28];
  const float* e_out_b = (const float*)d_in[29];
  const float* e_ffn_w = (const float*)d_in[30];
  const float* e_ffn_b = (const float*)d_in[31];
  const float* e_fc1_w = (const float*)d_in[32];
  const float* e_fc1_b = (const float*)d_in[33];
  const float* e_dw_w  = (const float*)d_in[34];
  const float* e_dw_b  = (const float*)d_in[35];
  const float* e_fc2_w = (const float*)d_in[36];
  const float* e_fc2_b = (const float*)d_in[37];

  // ---- outputs: [x (NX*C)] [xs (L*C)] [h (NL*L*DI*DS)] ----
  float* Xp   = (float*)d_out;
  float* XSp  = Xp + NX_ * C_;
  float* Hout = XSp + (size_t)L_ * C_;
  hipMemcpyAsync(Xp,  x_in,  sizeof(float) * NX_ * C_, hipMemcpyDeviceToDevice, stream);
  hipMemcpyAsync(XSp, xs_in, sizeof(float) * (size_t)L_ * C_, hipMemcpyDeviceToDevice, stream);

  // ---- workspace bump allocator ----
  float* wp = (float*)d_ws;
  auto alloc = [&](size_t n) { float* p = wp; wp += n; return p; };
  float* XSN = alloc((size_t)L_ * C_);
  float* XZ  = alloc((size_t)L_ * 2 * DI_);
  float* XC  = alloc((size_t)L_ * DI_);
  float* DBC = alloc((size_t)L_ * 64);
  float* DLT = alloc((size_t)L_ * DI_);
  float* Yb  = alloc((size_t)L_ * DI_);
  float* YG  = alloc((size_t)L_ * DI_);
  float* MO  = alloc((size_t)L_ * C_);
  float* FN  = alloc((size_t)L_ * C_);
  float* QN  = alloc((size_t)L_ * C_);
  float* QP  = alloc((size_t)L_ * C_);
  float* KP  = alloc((size_t)L_ * C_);
  float* VP  = alloc((size_t)L_ * C_);
  float* Sb  = alloc((size_t)L_ * NX_);   // max Mq*Nk = 1024*256 (or 256*1024)
  float* Ob  = alloc((size_t)L_ * C_);
  float* AO  = alloc((size_t)L_ * C_);
  float* LNF = alloc((size_t)L_ * C_);
  float* T1  = alloc((size_t)L_ * FH_);
  float* T2  = alloc((size_t)L_ * FH_);
  float* FO  = alloc((size_t)L_ * C_);

  auto ew = [](int n) { return dim3((n + 255) / 256); };
  auto gemm = [&](const float* A, int lda, const float* Bm, int ldb, int bT,
                  float* Cm, int ldc, int M, int N, int K, const float* bias, int act) {
    gemm_wmma_k<<<dim3(N / 64, M / 16), 32, 0, stream>>>(A, lda, Bm, ldb, bT, Cm, ldc, K,
                                                         bias, act);
  };

  // ---- multi-head cross attention -> AO[Mq, C] ----
  auto attn = [&](const float* Q, int Mq, const float* KV, int Nk,
                  const float* inw, const float* inb, const float* ow, const float* ob) {
    gemm(Q,  C_, inw,              C_, 0, QP, C_, Mq, C_, C_, inb,          0);
    gemm(KV, C_, inw + C_ * C_,    C_, 0, KP, C_, Nk, C_, C_, inb + C_,     0);
    gemm(KV, C_, inw + 2 * C_ * C_,C_, 0, VP, C_, Nk, C_, C_, inb + 2 * C_, 0);
    for (int hh = 0; hh < H_; ++hh) {
      gemm(QP + hh * DH_, C_, KP + hh * DH_, C_, 0, Sb, Nk, Mq, Nk, DH_, nullptr, 0);
      softmax_row_k<<<Mq, 32, 0, stream>>>(Sb, Nk, 0.125f);  // DH^-0.5
      gemm(Sb, Nk, VP + hh * DH_, C_, 1, Ob + hh * DH_, C_, Mq, DH_, Nk, nullptr, 0);
    }
    gemm(Ob, C_, ow, C_, 0, AO, C_, Mq, C_, C_, ob, 0);
  };

  // ---- Extractor block j: attn(q=xs, kv=x) + ConvFFN ----
  auto extractor = [&](int j) {
    ln_k<<<NX_, 32, 0, stream>>>(Xp,  e_fn_w + j * C_, e_fn_b + j * C_, FN, C_);
    ln_k<<<L_,  32, 0, stream>>>(XSp, e_qn_w + j * C_, e_qn_b + j * C_, QN, C_);
    attn(QN, L_, FN, NX_, e_in_w + (size_t)j * 3 * C_ * C_, e_in_b + (size_t)j * 3 * C_,
         e_out_w + (size_t)j * C_ * C_, e_out_b + (size_t)j * C_);
    add_k<<<ew(L_ * C_), 256, 0, stream>>>(XSp, AO, XSp, L_ * C_);
    ln_k<<<L_, 32, 0, stream>>>(XSp, e_ffn_w + j * C_, e_ffn_b + j * C_, LNF, C_);
    gemm(LNF, C_, e_fc1_w + (size_t)j * FH_ * C_, C_, 0, T1, FH_, L_, FH_, C_,
         e_fc1_b + (size_t)j * FH_, 0);
    dw3x3_gelu_k<<<ew(L_ * FH_), 256, 0, stream>>>(T1, e_dw_w + (size_t)j * FH_ * 9,
                                                   e_dw_b + (size_t)j * FH_, T2);
    gemm(T2, FH_, e_fc2_w + (size_t)j * C_ * FH_, FH_, 0, FO, C_, L_, C_, FH_,
         e_fc2_b + (size_t)j * C_, 0);
    add_k<<<ew(L_ * C_), 256, 0, stream>>>(XSp, FO, XSp, L_ * C_);
  };

  for (int i = 0; i < NL_; ++i) {
    const float* in_w   = m_in_w  + (size_t)i * 2 * DI_ * C_;
    const float* conv_w = m_conv_w+ (size_t)i * DI_ * 3;
    const float* conv_b = m_conv_b+ (size_t)i * DI_;
    const float* xp_w   = m_xp_w  + (size_t)i * (DR_ + 2 * DS_) * DI_;
    const float* dt_w   = m_dt_w  + (size_t)i * DI_ * DR_;
    const float* dt_b   = m_dt_b  + (size_t)i * DI_;
    const float* A_log  = m_A_log + (size_t)i * DI_ * DS_;
    const float* Dv     = m_D     + (size_t)i * DI_;
    const float* out_w  = m_out_w + (size_t)i * C_ * DI_;

    // ----- Mamba block -----
    rms_k<<<L_, 32, 0, stream>>>(XSp, m_norm_w + i * C_, XSN, C_);
    gemm(XSN, C_, in_w, C_, 0, XZ, 2 * DI_, L_, 2 * DI_, C_, nullptr, 0);       // in_proj
    conv1d_silu_k<<<ew(L_ * DI_), 256, 0, stream>>>(XZ, 2 * DI_, conv_w, conv_b, XC);
    gemm(XC, DI_, xp_w, DI_, 0, DBC, 64, L_, 64, DI_, nullptr, 0);              // x_proj
    gemm(DBC, 64, dt_w, DR_, 0, DLT, DI_, L_, DI_, DR_, dt_b, 3);               // dt + softplus
    scan_k<<<ew(L_ * DI_), 256, 0, stream>>>(XC, DLT, DBC, A_log, Dv,
                                             h_in + (size_t)i * L_ * DI_ * DS_,
                                             Hout + (size_t)i * L_ * DI_ * DS_, Yb);
    gate_silu_k<<<ew(L_ * DI_), 256, 0, stream>>>(XZ + DI_, 2 * DI_, Yb, YG);
    gemm(YG, DI_, out_w, DI_, 0, MO, C_, L_, C_, DI_, nullptr, 0);              // out_proj
    add_k<<<ew(L_ * C_), 256, 0, stream>>>(MO, XSN, XSp, L_ * C_);              // xs = out + xs_n

    // ----- Injector: x += gamma * MHA(ln(x), ln(xs)) -----
    ln_k<<<L_,  32, 0, stream>>>(XSp, i_fn_w + i * C_, i_fn_b + i * C_, FN, C_);
    ln_k<<<NX_, 32, 0, stream>>>(Xp,  i_qn_w + i * C_, i_qn_b + i * C_, QN, C_);
    attn(QN, NX_, FN, L_, i_in_w + (size_t)i * 3 * C_ * C_, i_in_b + (size_t)i * 3 * C_,
         i_out_w + (size_t)i * C_ * C_, i_out_b + (size_t)i * C_);
    axpy_col_k<<<ew(NX_ * C_), 256, 0, stream>>>(Xp, i_gamma + i * C_, AO, NX_ * C_, C_);

    // ----- Extractor(s) -----
    extractor(i);
    if (i == NL_ - 1) { extractor(4); extractor(5); }
  }
}
